// VQuantized_39230231281715
// MI455X (gfx1250) — matrix operand order repaired and checked by reading, not compiled, not run on hardware
//
#include <hip/hip_runtime.h>

// ---------------------------------------------------------------------------
// VQ quantization for MI455X (gfx1250, wave32, WMMA)
//   dist argmin via  argmin_k ( |e_k|^2 - 2 * x . e_k )   (|x|^2 dropped)
//   cross term computed with v_wmma_f32_16x16x32_bf16, f32 accumulate.
// ---------------------------------------------------------------------------

typedef __attribute__((ext_vector_type(16))) __bf16 v16bf;
typedef __attribute__((ext_vector_type(8)))  __bf16 v8bf;
typedef __attribute__((ext_vector_type(8)))  float  v8f;

#define NUM_CODES 1024
#define DIM       256
#define HW        4096          // 64*64, rows per batch image
#define NROWS     131072        // 32*64*64
#define MT        64            // rows per workgroup (4 waves x 16 rows)
#define LDA       264           // padded bf16 stride for A tile in LDS
#define LDB       272           // padded bf16 stride for B tile in LDS
#define CHUNK     32            // codes staged per LDS chunk

// --------------------------- kernel 0: prep --------------------------------
// codebook f32 -> bf16 (for WMMA B operand) and |e_k|^2 per code.
__global__ __launch_bounds__(256)
void vq_prep_kernel(const float* __restrict__ cb,
                    __bf16* __restrict__ cbbf,
                    float* __restrict__ enorm) {
    __shared__ float sred[8];
    const int k = blockIdx.x;
    const int t = threadIdx.x;
    float v = cb[(size_t)k * DIM + t];
    cbbf[(size_t)k * DIM + t] = (__bf16)v;
    float sq = v * v;
#pragma unroll
    for (int off = 16; off >= 1; off >>= 1)
        sq += __shfl_down(sq, off, 32);
    if ((t & 31) == 0) sred[t >> 5] = sq;
    __syncthreads();
    if (t < 8) {
        float s = sred[t];
#pragma unroll
        for (int off = 4; off >= 1; off >>= 1)
            s += __shfl_down(s, off, 8);
        if (t == 0) enorm[k] = s;
    }
}

// --------------------------- kernel 1: argmin ------------------------------
// Each block: 64 consecutive rows (same batch image), 4 waves x 16-row strips.
__global__ __launch_bounds__(128)
void vq_argmin_kernel(const float* __restrict__ x,
                      const __bf16* __restrict__ cbbf,
                      const float* __restrict__ enorm,
                      int* __restrict__ idx_out) {
    __shared__ __align__(16) __bf16 sA[MT * LDA];      // 33792 B
    __shared__ __align__(16) __bf16 sB[CHUNK * LDB];   // 17408 B
    __shared__ float sE[NUM_CODES];                    //  4096 B

    const int tid  = threadIdx.x;
    const int base = blockIdx.x * MT;      // first global row of tile
    const int b    = base >> 12;           // batch image (4096 rows each)
    const int hw0  = base & 4095;
    const float* xb = x + (size_t)b * DIM * HW + hw0;

    // Stage A: transpose NCHW -> [m][c], convert to bf16. Coalesced: for each
    // channel c, the 64 row-values are contiguous in global memory.
    for (int i = tid; i < MT * DIM; i += 128) {
        const int m = i & (MT - 1);
        const int c = i >> 6;
        sA[m * LDA + c] = (__bf16)xb[(size_t)c * HW + m];
    }
    for (int i = tid; i < NUM_CODES; i += 128) sE[i] = enorm[i];
    __syncthreads();

    const int lane = tid & 31;
    const int wave = tid >> 5;
    const int half = lane >> 4;
    const int l16  = lane & 15;
    const int mrow = wave * 16 + l16;

    // A fragments for this wave's 16-row strip, all 8 K-tiles kept resident.
    // ISA layout: lanes 0-15 K=[0..7]+[16..23], lanes 16-31 K=[8..15]+[24..31].
    union BF { v16bf v; v8bf h[2]; };
    BF afrag[8];
#pragma unroll
    for (int kt = 0; kt < 8; ++kt) {
        const __bf16* ap = &sA[mrow * LDA + kt * 32 + half * 8];
        afrag[kt].h[0] = *(const v8bf*)(ap);
        afrag[kt].h[1] = *(const v8bf*)(ap + 16);
    }

    float dmin[8];
    int   imin[8];
#pragma unroll
    for (int v = 0; v < 8; ++v) { dmin[v] = 3.4e38f; imin[v] = 0; }

    for (int ch = 0; ch < NUM_CODES / CHUNK; ++ch) {
        __syncthreads();   // all waves done reading previous chunk
        // Stage 32 codes x 256 dims of bf16 codebook (16 KB) into LDS.
#pragma unroll
        for (int k = 0; k < 8; ++k) {
            const int ci   = tid + k * 128;   // 16-byte chunk id (1024 total)
            const int code = ci >> 5;         // 32 chunks per code row
            const int coff = (ci & 31) * 8;
            *(v8bf*)&sB[code * LDB + coff] =
                *(const v8bf*)&cbbf[(size_t)(ch * CHUNK + code) * DIM + coff];
        }
        __syncthreads();

#pragma unroll
        for (int sub = 0; sub < 2; ++sub) {
            const int n0 = ch * CHUNK + sub * 16;
            v8f acc = {};
#pragma unroll
            for (int kt = 0; kt < 8; ++kt) {
                // B layout: lane holds code column (l16); lanes 0-15
                // K=[0..7]+[16..23], lanes 16-31 K=[8..15]+[24..31] (matches A).
                BF bfrag;
                const __bf16* bp =
                    &sB[(sub * 16 + l16) * LDB + kt * 32 + half * 8];
                bfrag.h[0] = *(const v8bf*)(bp);
                bfrag.h[1] = *(const v8bf*)(bp + 16);
                acc = __builtin_amdgcn_wmma_f32_16x16x32_bf16(
                    false, afrag[kt].v, false, bfrag.v,
                    (short)0, acc, false, false);
            }
            const float en2  = sE[n0 + l16];
            const int   nidx = n0 + l16;
#pragma unroll
            for (int v = 0; v < 8; ++v) {
                const float d = en2 - 2.0f * acc[v];
                if (d < dmin[v]) { dmin[v] = d; imin[v] = nidx; }
            }
        }
    }

    // C/D layout: lanes 0-15 hold rows v, lanes 16-31 hold rows v+8; N = l16.
    // Argmin across the 16 lanes of each half (tie -> lowest index).
#pragma unroll
    for (int v = 0; v < 8; ++v) {
        float dv = dmin[v];
        int   iv = imin[v];
#pragma unroll
        for (int off = 8; off >= 1; off >>= 1) {
            const float od = __shfl_xor(dv, off, 16);
            const int   oi = __shfl_xor(iv, off, 16);
            if (od < dv || (od == dv && oi < iv)) { dv = od; iv = oi; }
        }
        if (l16 == 0) {
            const int row = base + wave * 16 + v + 8 * half;
            idx_out[row] = iv;
        }
    }
}

// --------------------------- kernel 2: gather ------------------------------
// quantized[b,c,h,w] = codebook[idx[n], c]; writes coalesced in NCHW.
__global__ __launch_bounds__(128)
void vq_gather_kernel(const float* __restrict__ cb,
                      const int* __restrict__ idx,
                      float* __restrict__ qout) {
    __shared__ int sIdx[MT];
    const int tid  = threadIdx.x;
    const int base = blockIdx.x * MT;
    const int b    = base >> 12;
    const int hw0  = base & 4095;
    if (tid < MT) sIdx[tid] = idx[base + tid];
    __syncthreads();
    float* ob = qout + (size_t)b * DIM * HW + hw0;
    for (int i = tid; i < MT * DIM; i += 128) {
        const int m = i & (MT - 1);
        const int c = i >> 6;
        ob[(size_t)c * HW + m] = cb[(size_t)sIdx[m] * DIM + c];
    }
}

// --------------------------- launcher --------------------------------------
extern "C" void kernel_launch(void* const* d_in, const int* in_sizes, int n_in,
                              void* d_out, int out_size, void* d_ws, size_t ws_size,
                              hipStream_t stream) {
    const float* x  = (const float*)d_in[0];   // (32,256,64,64) f32
    const float* cb = (const float*)d_in[1];   // (1024,256) f32

    __bf16* cbbf  = (__bf16*)d_ws;                                   // 512 KB
    float*  enorm = (float*)((char*)d_ws + NUM_CODES * DIM * 2);     //   4 KB

    int*   idx_out = (int*)d_out;                   // first N elements
    float* qout    = (float*)d_out + NROWS;         // then B*C*H*W floats

    vq_prep_kernel<<<NUM_CODES, 256, 0, stream>>>(cb, cbbf, enorm);
    vq_argmin_kernel<<<NROWS / MT, 128, 0, stream>>>(x, cbbf, enorm, idx_out);
    vq_gather_kernel<<<NROWS / MT, 128, 0, stream>>>(cb, idx_out, qout);
}
